// Directed_Bipartite_MH_Attention_1606317769006
// MI455X (gfx1250) — matrix operand order repaired and checked by reading, not compiled, not run on hardware
//
#include <hip/hip_runtime.h>

// ---------------------------------------------------------------------------
// Directed bipartite multi-head attention for MI455X (gfx1250, wave32, WMMA)
//
//   prep1 : gather+convert q/k/v to bf16, transpose W* to bf16, zero ao
//   prep2 : scatter head rows of query into ao (bf16)
//   pack  : pkT[H][T] u8 = (c_indices & 63) | (adj ? 64 : 0)   (transposed)
//   gemm  : Qp = (qg @ WqT + bq)*(scale*log2e)  [T][256] bf16
//           Kt = (kg @ WkT + bk)^T              [256][H] bf16
//           Vp = (vg @ WvT + bv)                [H][256] bf16
//   attn  : flash-style base-2 online softmax; 8 waves = 8 heads per block;
//           bias LUT fed as WMMA C operand; scores transposed through LDS so
//           max/sum are register trees + one xor-16 shuffle; P packed into
//           the A-fragment in registers; corr/1-over-sum broadcast via LDS;
//           2x-unrolled ping-pong pipeline (prefetch distance 2, no copies)
//   gemm  : out = ao @ WoT + bo                 [N][256] f32
// ---------------------------------------------------------------------------

#define T_NODES 6000
#define H_NODES 4000
#define N_NODES 12000
#define DM      256
#define NH      8
#define HD      32
#define EVOC    64
#define STW     36   // padded LDS row stride (floats) for score transpose

typedef __attribute__((ext_vector_type(16))) __bf16 v16bf;
typedef __attribute__((ext_vector_type(8)))  __bf16 v8bf;
typedef __attribute__((ext_vector_type(8)))  float  v8f;
typedef __attribute__((ext_vector_type(4)))  float  v4f;
typedef unsigned long long u64;

#define CAT16(lo, hi) __builtin_shufflevector((lo), (hi), 0,1,2,3,4,5,6,7,8,9,10,11,12,13,14,15)

static __device__ __forceinline__ v8f wmma_bf16(v16bf a, v16bf b, v8f c) {
  return __builtin_amdgcn_wmma_f32_16x16x32_bf16(false, a, false, b, (short)0, c, false, false);
}

static __device__ __forceinline__ float fast_exp2(float x) {
#if __has_builtin(__builtin_amdgcn_exp2f)
  return __builtin_amdgcn_exp2f(x);
#else
  return __expf(x * 0.6931471805599453f);
#endif
}

static __device__ __forceinline__ float vmax4(v4f v) {
  return fmaxf(fmaxf(v[0], v[1]), fmaxf(v[2], v[3]));
}

// ---------------------------------------------------------------------------
// prep1: elementwise gathers/converts/transposes + zero ao
// ---------------------------------------------------------------------------
__global__ __launch_bounds__(256) void prep1_kernel(
    const float* __restrict__ query, const float* __restrict__ key,
    const float* __restrict__ value,
    const int* __restrict__ git, const int* __restrict__ gih,
    const float* __restrict__ Wq, const float* __restrict__ Wk,
    const float* __restrict__ Wv, const float* __restrict__ Wo,
    __bf16* __restrict__ qg, __bf16* __restrict__ kg, __bf16* __restrict__ vg,
    __bf16* __restrict__ WqT, __bf16* __restrict__ WkT,
    __bf16* __restrict__ WvT, __bf16* __restrict__ WoT,
    __bf16* __restrict__ ao)
{
  const int QG = T_NODES * DM;
  const int KG = H_NODES * DM;
  const int WT4 = 4 * DM * DM;
  const int AO = N_NODES * DM;
  int i = blockIdx.x * 256 + threadIdx.x;
  const int total = QG + 2 * KG + WT4 + AO;
  if (i >= total) return;

  if (i < QG) {
    int t = i >> 8, d = i & 255;
    qg[i] = (__bf16)query[git[t] * DM + d];
  } else if (i < QG + KG) {
    int e = i - QG;
    int j = e >> 8, d = e & 255;
    kg[e] = (__bf16)key[gih[j] * DM + d];
  } else if (i < QG + 2 * KG) {
    int e = i - QG - KG;
    int j = e >> 8, d = e & 255;
    vg[e] = (__bf16)value[gih[j] * DM + d];
  } else if (i < QG + 2 * KG + WT4) {
    int e = i - QG - 2 * KG;
    int w = e >> 16;
    int t = e & 65535;
    int d = t >> 8, j = t & 255;
    float val;
    __bf16* dst;
    if      (w == 0) { val = Wq[j * DM + d]; dst = WqT; }
    else if (w == 1) { val = Wk[j * DM + d]; dst = WkT; }
    else if (w == 2) { val = Wv[j * DM + d]; dst = WvT; }
    else             { val = Wo[j * DM + d]; dst = WoT; }
    dst[t] = (__bf16)val;
  } else {
    int e = i - QG - 2 * KG - WT4;
    ao[e] = (__bf16)0.0f;
  }
}

__global__ __launch_bounds__(256) void prep2_kernel(
    const float* __restrict__ query, const int* __restrict__ gih,
    __bf16* __restrict__ ao)
{
  int i = blockIdx.x * 256 + threadIdx.x;
  if (i >= H_NODES * DM) return;
  int j = i >> 8, d = i & 255;
  int row = gih[j];
  ao[row * DM + d] = (__bf16)query[row * DM + d];
}

// ---------------------------------------------------------------------------
// pack: pkT[h][t] = (cidx[t][h] & 63) | (adj[t][h] ? 64 : 0)
// ---------------------------------------------------------------------------
__global__ __launch_bounds__(256) void pack_kernel(
    const int* __restrict__ adj, const int* __restrict__ cidx,
    unsigned char* __restrict__ pkT)
{
  __shared__ unsigned char tile[64][68];
  const int tb = blockIdx.x * 64;  // t base
  const int hb = blockIdx.y * 64;  // h base

  const int hh = threadIdx.x & 63;
  const int t0 = threadIdx.x >> 6;
#pragma unroll
  for (int tt = t0; tt < 64; tt += 4) {
    int t = tb + tt, h = hb + hh;
    unsigned char v = 0;
    if (t < T_NODES && h < H_NODES) {
      size_t idx = (size_t)t * H_NODES + h;
      int a = adj[idx];
      int c = cidx[idx];
      v = (unsigned char)((c & 63) | (a ? 64 : 0));
    }
    tile[hh][tt] = v;
  }
  __syncthreads();

  for (int w = threadIdx.x; w < 1024; w += 256) {
    int hh2 = w >> 4, tw = (w & 15) * 4;
    int h = hb + hh2, t = tb + tw;
    if (h < H_NODES && t + 3 < T_NODES) {
      unsigned int word;
      __builtin_memcpy(&word, &tile[hh2][tw], 4);
      *(unsigned int*)(pkT + (size_t)h * T_NODES + t) = word;
    }
  }
}

// ---------------------------------------------------------------------------
// bf16 GEMM: C[M][256] = (A[M][256] @ WT[256][256] + bias) * scale
// ---------------------------------------------------------------------------
template <bool TRANS_OUT, typename OutT>
__global__ __launch_bounds__(256) void gemm_bf16_kernel(
    const __bf16* __restrict__ A, const __bf16* __restrict__ WT,
    const float* __restrict__ bias, OutT* __restrict__ C,
    int M, float scale)
{
  const int lane = threadIdx.x & 31;
  const int w = blockIdx.x * 8 + (threadIdx.x >> 5);
  const int mt = w >> 4, nt = w & 15;
  const int mb = mt * 16, nb = nt * 16;

  const int arow = mb + (lane & 15);
  const int k0 = (lane >> 4) * 8;

  v8f acc = {};
#pragma unroll
  for (int kc = 0; kc < DM; kc += 32) {
    v8bf alo = *(const v8bf*)(A + arow * DM + kc + k0);
    v8bf ahi = *(const v8bf*)(A + arow * DM + kc + k0 + 16);
    v16bf a = CAT16(alo, ahi);
    v16bf b;
#pragma unroll
    for (int s = 0; s < 16; ++s)
      b[s] = WT[(kc + k0 + (s & 7) + ((s >> 3) << 4)) * DM + nb + (lane & 15)];
    acc = wmma_bf16(a, b, acc);
  }

  const int n = nb + (lane & 15);
  const float bv = bias[n];
  const int half8 = (lane >> 4) * 8;

  if (TRANS_OUT) {
    v8bf tmp;
#pragma unroll
    for (int r = 0; r < 8; ++r) tmp[r] = (__bf16)((acc[r] + bv) * scale);
    *(v8bf*)((__bf16*)C + (size_t)n * M + mb + half8) = tmp;
  } else {
#pragma unroll
    for (int r = 0; r < 8; ++r)
      C[(size_t)(mb + half8 + r) * DM + n] = (OutT)((acc[r] + bv) * scale);
  }
}

// ---------------------------------------------------------------------------
// Attention
// ---------------------------------------------------------------------------
struct Chunk {
  v16bf kb0, kb1;   // K B-fragments (two 16-key subtiles)
  v16bf vb0, vb1;   // V B-fragments (out cols 0-15, 16-31)
  u64   pk0, pk1;   // packed mask/edge bytes, 8 query rows each
};

__global__ __launch_bounds__(256) void attn_kernel(
    const __bf16* __restrict__ Qp,   // [T][256] (scale*log2e folded in)
    const __bf16* __restrict__ Kt,   // [256][H]
    const __bf16* __restrict__ Vp,   // [H][256]
    const unsigned char* __restrict__ pkT,  // [H][T]
    const float* __restrict__ edge,  // [64][8]
    const int* __restrict__ git,     // [T]
    __bf16* __restrict__ ao)         // [N][256]
{
  __shared__ __attribute__((aligned(32))) float s_bias[128 * NH];   // 4KB LUT
  __shared__ __attribute__((aligned(16))) float s_t[8][16 * STW];   // 18KB
  __shared__ __attribute__((aligned(16))) float s_c[8][16];         // 512B

  const int tid  = threadIdx.x;
  const int wave = tid >> 5;        // head index
  const int lane = tid & 31;
  const int h    = wave;
  const int tb   = blockIdx.x * 16; // query tile base

  // bias LUT: idx = packed(7b)*8 + head; premultiplied by log2(e)
  for (int i = tid; i < 128 * NH; i += 256) {
    int p = i >> 3, hh = i & 7;
    s_bias[i] = (edge[(p & 63) * NH + hh] + ((p & 64) ? 0.0f : -1.0e9f))
              * 1.4426950408889634f;
  }
  __syncthreads();

  const int half = lane >> 4;       // 0 | 1
  const int ncol = lane & 15;       // D-layout column; also my owned row
  const int k0   = half * 8;

  // Q A-fragment (reused every iteration)
  const __bf16* qrow = Qp + (size_t)(tb + ncol) * DM + h * HD;
  v8bf qlo = *(const v8bf*)(qrow + k0);
  v8bf qhi = *(const v8bf*)(qrow + k0 + 16);
  v16bf qa = CAT16(qlo, qhi);

  const __bf16* krow = Kt + (size_t)(h * HD) * H_NODES + ncol;
  const unsigned char* prow = pkT + tb + 8 * half;

  auto load_chunk = [&](int kv) {
    if (kv >= H_NODES) kv = 0;   // clamped prefetch beyond the end
    Chunk c;
    c.pk0 = *(const u64*)(prow + (size_t)(kv + ncol) * T_NODES);
    c.pk1 = *(const u64*)(prow + (size_t)(kv + 16 + ncol) * T_NODES);
    const __bf16* vrow = Vp + (size_t)kv * DM + h * HD + ncol;
#pragma unroll
    for (int s = 0; s < 16; ++s) {
      int kk = k0 + (s & 7) + ((s >> 3) << 4);
      c.kb0[s] = krow[(size_t)kk * H_NODES + kv];
      c.kb1[s] = krow[(size_t)kk * H_NODES + kv + 16];
      c.vb0[s] = vrow[(size_t)kk * DM];
      c.vb1[s] = vrow[(size_t)kk * DM + 16];
    }
    return c;
  };

  v8f acc0 = {}, acc1 = {};
  float rm = -3.0e38f;   // running max of my row (row = ncol)
  float rs = 0.0f;       // running sum of my row

  float* st = s_t[wave];
  const float* rowp = st + ncol * STW;
  float* cp = s_c[wave];

  auto process = [&](const Chunk& c) {
    // bias tiles in C/D layout (bias folded into WMMA C operand)
    v8f c0, c1;
#pragma unroll
    for (int r = 0; r < 8; ++r) {
      unsigned b0 = ((unsigned)(c.pk0 >> (8 * r)) & 127u);
      unsigned b1 = ((unsigned)(c.pk1 >> (8 * r)) & 127u);
      c0[r] = s_bias[(b0 << 3) + h];
      c1[r] = s_bias[(b1 << 3) + h];
    }

    // scores = Q.K + bias (log2 domain)
    v8f s0 = wmma_bf16(qa, c.kb0, c0);
    v8f s1 = wmma_bf16(qa, c.kb1, c1);

    // transpose scores through LDS: each lane then owns row `ncol`,
    // k slots (k0..k0+7, k0+16..k0+23) -- exactly its A-fragment slots
#pragma unroll
    for (int r = 0; r < 8; ++r) {
      int m = r + half * 8;
      st[m * STW + ncol]      = s0[r];
      st[m * STW + 16 + ncol] = s1[r];
    }
    v4f x0 = *(const v4f*)(rowp + k0);
    v4f x1 = *(const v4f*)(rowp + k0 + 4);
    v4f x2 = *(const v4f*)(rowp + k0 + 16);
    v4f x3 = *(const v4f*)(rowp + k0 + 20);

    // row max: register tree + one xor-16 shuffle (partner holds other ks)
    float mx = fmaxf(fmaxf(vmax4(x0), vmax4(x1)), fmaxf(vmax4(x2), vmax4(x3)));
    mx = fmaxf(mx, __shfl_xor(mx, 16));
    float nm = fmaxf(rm, mx);
    float corr = fast_exp2(rm - nm);
    rm = nm;

    // exp + pack P directly into A-fragment order + row sum
    v16bf pa;
    float ts = 0.0f;
#pragma unroll
    for (int j = 0; j < 4; ++j) {
      float e0 = fast_exp2(x0[j] - nm);
      float e1 = fast_exp2(x1[j] - nm);
      float e2 = fast_exp2(x2[j] - nm);
      float e3 = fast_exp2(x3[j] - nm);
      pa[j]      = (__bf16)e0;
      pa[4 + j]  = (__bf16)e1;
      pa[8 + j]  = (__bf16)e2;
      pa[12 + j] = (__bf16)e3;
      ts += (e0 + e1) + (e2 + e3);
    }
    ts += __shfl_xor(ts, 16);
    rs = rs * corr + ts;

    // broadcast per-row correction into D layout via LDS, rescale accs
    cp[ncol] = corr;   // both halves write the same value: benign
    v4f cva = *(const v4f*)(cp + 8 * half);
    v4f cvb = *(const v4f*)(cp + 8 * half + 4);
#pragma unroll
    for (int r = 0; r < 4; ++r) {
      acc0[r]     *= cva[r];
      acc1[r]     *= cva[r];
      acc0[4 + r] *= cvb[r];
      acc1[4 + r] *= cvb[r];
    }

    acc0 = wmma_bf16(pa, c.vb0, acc0);
    acc1 = wmma_bf16(pa, c.vb1, acc1);
  };

  // ping-pong pipeline, prefetch distance 2, no chunk register copies.
  // 125 chunks of 32 keys = 62 double iterations + 1 peeled tail.
  Chunk a = load_chunk(0);
  Chunk b = load_chunk(32);
  for (int kv = 0; kv < H_NODES - 64; kv += 64) {
    process(a);
    a = load_chunk(kv + 64);
    process(b);
    b = load_chunk(kv + 96);
  }
  process(a);   // chunk H_NODES-32; b holds a discarded clamped prefetch

  // epilogue: normalize (broadcast 1/rowsum via LDS), scatter rows
  cp[ncol] = 1.0f / rs;
  v4f iva = *(const v4f*)(cp + 8 * half);
  v4f ivb = *(const v4f*)(cp + 8 * half + 4);
#pragma unroll
  for (int r = 0; r < 8; ++r) {
    int m = r + half * 8;
    int orow = git[tb + m];
    float t = (r < 4) ? iva[r & 3] : ivb[r & 3];
    ao[(size_t)orow * DM + h * HD + ncol]      = (__bf16)(acc0[r] * t);
    ao[(size_t)orow * DM + h * HD + 16 + ncol] = (__bf16)(acc1[r] * t);
  }
}

// ---------------------------------------------------------------------------
extern "C" void kernel_launch(void* const* d_in, const int* in_sizes, int n_in,
                              void* d_out, int out_size, void* d_ws, size_t ws_size,
                              hipStream_t stream) {
  const float* query = (const float*)d_in[0];
  const float* key   = (const float*)d_in[1];
  const float* value = (const float*)d_in[2];
  const int*   adj   = (const int*)d_in[3];
  const int*   cidx  = (const int*)d_in[4];
  const int*   git   = (const int*)d_in[5];
  const int*   gih   = (const int*)d_in[6];
  const float* Wq = (const float*)d_in[7];
  const float* bq = (const float*)d_in[8];
  const float* Wk = (const float*)d_in[9];
  const float* bk = (const float*)d_in[10];
  const float* Wv = (const float*)d_in[11];
  const float* bv = (const float*)d_in[12];
  const float* Wo = (const float*)d_in[13];
  const float* bo = (const float*)d_in[14];
  const float* edge = (const float*)d_in[15];
  float* out = (float*)d_out;

  size_t off = 0;
  auto carve = [&](size_t bytes) {
    void* p = (char*)d_ws + off;
    off = (off + bytes + 255) & ~(size_t)255;
    return p;
  };
  __bf16* qg  = (__bf16*)carve((size_t)T_NODES * DM * 2);
  __bf16* kg  = (__bf16*)carve((size_t)H_NODES * DM * 2);
  __bf16* vg  = (__bf16*)carve((size_t)H_NODES * DM * 2);
  __bf16* WqT = (__bf16*)carve((size_t)DM * DM * 2);
  __bf16* WkT = (__bf16*)carve((size_t)DM * DM * 2);
  __bf16* WvT = (__bf16*)carve((size_t)DM * DM * 2);
  __bf16* WoT = (__bf16*)carve((size_t)DM * DM * 2);
  __bf16* Qp  = (__bf16*)carve((size_t)T_NODES * DM * 2);
  __bf16* Kt  = (__bf16*)carve((size_t)DM * H_NODES * 2);
  __bf16* Vp  = (__bf16*)carve((size_t)H_NODES * DM * 2);
  __bf16* ao  = (__bf16*)carve((size_t)N_NODES * DM * 2);
  unsigned char* pkT = (unsigned char*)carve((size_t)H_NODES * T_NODES);

  {
    int total = T_NODES * DM + 2 * H_NODES * DM + 4 * DM * DM + N_NODES * DM;
    prep1_kernel<<<(total + 255) / 256, 256, 0, stream>>>(
        query, key, value, git, gih, Wq, Wk, Wv, Wo,
        qg, kg, vg, WqT, WkT, WvT, WoT, ao);
    prep2_kernel<<<(H_NODES * DM + 255) / 256, 256, 0, stream>>>(query, gih, ao);
  }

  {
    dim3 grid((T_NODES + 63) / 64, (H_NODES + 63) / 64);
    pack_kernel<<<grid, 256, 0, stream>>>(adj, cidx, pkT);
  }

  // scale = 1/sqrt(32) * log2(e): softmax runs in base-2 domain
  const float scaling = 0.17677669529663687f * 1.4426950408889634f;

  gemm_bf16_kernel<false, __bf16><<<T_NODES / 8, 256, 0, stream>>>(
      qg, WqT, bq, Qp, T_NODES, scaling);
  gemm_bf16_kernel<true, __bf16><<<H_NODES / 8, 256, 0, stream>>>(
      kg, WkT, bk, Kt, H_NODES, 1.0f);
  gemm_bf16_kernel<false, __bf16><<<H_NODES / 8, 256, 0, stream>>>(
      vg, WvT, bv, Vp, H_NODES, 1.0f);

  attn_kernel<<<T_NODES / 16, 256, 0, stream>>>(
      Qp, Kt, Vp, pkT, edge, git, ao);

  gemm_bf16_kernel<false, float><<<N_NODES / 8, 256, 0, stream>>>(
      ao, WoT, bo, out, N_NODES, 1.0f);
}
